// SelfAttentionMultiHead_18940805776011
// MI455X (gfx1250) — matrix operand order repaired and checked by reading, not compiled, not run on hardware
//
#include <hip/hip_runtime.h>

typedef __bf16 bf16_t;
typedef __attribute__((ext_vector_type(16))) bf16_t v16bf;
typedef __attribute__((ext_vector_type(8)))  bf16_t v8bf;
typedef __attribute__((ext_vector_type(8)))  float  v8f;

union FragA { v16bf v; v8bf h[2]; };

#define DIMX     1024
#define HEADS    16
#define HEAD_DIM 64
#define SEQ      2048
#define BATCH    4
#define MTOT     (BATCH * SEQ)   // 8192 token rows
#define QKV_N    (3 * DIMX)      // 3072
#define ATT_SCALE 0.03125f       // DIM^-0.5 (full dim per reference)

// ---------------------------------------------------------------- fp32 -> bf16
__global__ void f32_to_bf16_kernel(const float* __restrict__ in,
                                   bf16_t* __restrict__ out, int n) {
    int i = blockIdx.x * blockDim.x + threadIdx.x;
    int stride = gridDim.x * blockDim.x;
    for (; i < n; i += stride) out[i] = (bf16_t)in[i];
}

// ---------------------------------------------------------------- QKV GEMM
// C[m, e] = sum_k X[m,k] * Wqkv[e,k];  M=8192, N=3072, K=1024
// wave tile 64x64, scatter to Q[bh,n,d], K[bh,n,d], Vt[bh,d,n] (bf16)
__global__ __launch_bounds__(128) void qkv_gemm_kernel(
    const bf16_t* __restrict__ X, const bf16_t* __restrict__ W,
    bf16_t* __restrict__ Qo, bf16_t* __restrict__ Ko, bf16_t* __restrict__ Vt) {
    const int lane  = threadIdx.x & 31;
    const int wave  = threadIdx.x >> 5;
    const int tileM = blockIdx.x * 64;
    const int tileN = blockIdx.y * 256 + wave * 64;
    const int half  = lane >> 4;
    const int l16   = lane & 15;
    const int base8  = half * 8;
    const int base16 = half * 16;

    v8f acc[4][4] = {};
    for (int k = 0; k < DIMX; k += 32) {
        FragA a[4];
#pragma unroll
        for (int mi = 0; mi < 4; ++mi) {
            const bf16_t* p = X + (size_t)(tileM + mi * 16 + l16) * DIMX + k + base16;
            a[mi].h[0] = *(const v8bf*)p;
            a[mi].h[1] = *(const v8bf*)(p + 8);
        }
        v16bf b[4];
#pragma unroll
        for (int ni = 0; ni < 4; ++ni) {
            const bf16_t* p = W + (size_t)(tileN + ni * 16 + l16) * DIMX + k + base16;
            b[ni] = *(const v16bf*)p;
        }
#pragma unroll
        for (int mi = 0; mi < 4; ++mi)
#pragma unroll
            for (int ni = 0; ni < 4; ++ni)
                acc[mi][ni] = __builtin_amdgcn_wmma_f32_16x16x32_bf16(
                    false, a[mi].v, false, b[ni], (short)0, acc[mi][ni], false, false);
    }
    // scatter C-layout results: row = tileM+mi*16+r+half*8, col e = tileN+ni*16+l16
#pragma unroll
    for (int mi = 0; mi < 4; ++mi) {
#pragma unroll
        for (int ni = 0; ni < 4; ++ni) {
#pragma unroll
            for (int r = 0; r < 8; ++r) {
                int m = tileM + mi * 16 + r + half * 8;
                int e = tileN + ni * 16 + l16;
                int b  = m >> 11;          // / SEQ
                int n  = m & (SEQ - 1);
                int s  = e >> 10;          // q/k/v selector
                int el = e & (DIMX - 1);
                int h  = el >> 6;
                int d  = el & (HEAD_DIM - 1);
                size_t bh = (size_t)(b * HEADS + h);
                bf16_t val = (bf16_t)acc[mi][ni][r];
                if (s == 0)      Qo[(bh * SEQ + n) * HEAD_DIM + d] = val;
                else if (s == 1) Ko[(bh * SEQ + n) * HEAD_DIM + d] = val;
                else             Vt[(bh * HEAD_DIM + d) * SEQ + n] = val;
            }
        }
    }
}

// ---------------------------------------------------------------- flash attention
// grid: (SEQ/64, BATCH*HEADS), block 128 (4 waves). Wave owns 16 query rows.
__global__ __launch_bounds__(128) void attn_kernel(
    const bf16_t* __restrict__ Q, const bf16_t* __restrict__ K,
    const bf16_t* __restrict__ Vt, bf16_t* __restrict__ AO) {
    __shared__ bf16_t plds[4][16][64];   // per-wave private 16x64 P tile
    const int lane  = threadIdx.x & 31;
    const int wave  = threadIdx.x >> 5;
    const int bh    = blockIdx.y;
    const int qbase = blockIdx.x * 64 + wave * 16;
    const int half  = lane >> 4;
    const int l16   = lane & 15;
    const int base8  = half * 8;
    const int base16 = half * 16;

    const bf16_t* Qp = Q  + (size_t)bh * SEQ * HEAD_DIM;
    const bf16_t* Kp = K  + (size_t)bh * SEQ * HEAD_DIM;
    const bf16_t* Vp = Vt + (size_t)bh * HEAD_DIM * SEQ;

    // Q A-fragments for the two 32-wide k-steps over head_dim
    FragA qa[2];
    {
        const bf16_t* p = Qp + (size_t)(qbase + l16) * HEAD_DIM;
        qa[0].h[0] = *(const v8bf*)(p + base16);
        qa[0].h[1] = *(const v8bf*)(p + base16 + 8);
        qa[1].h[0] = *(const v8bf*)(p + 32 + base16);
        qa[1].h[1] = *(const v8bf*)(p + 32 + base16 + 8);
    }

    float mrow[8], lrow[8];
    v8f o[4] = {};
#pragma unroll
    for (int r = 0; r < 8; ++r) { mrow[r] = -1e30f; lrow[r] = 0.f; }

    for (int jb = 0; jb < SEQ; jb += 64) {
        // S = Q @ K^T for 16 rows x 64 keys
        v8f s[4] = {};
#pragma unroll
        for (int jf = 0; jf < 4; ++jf) {
            const bf16_t* kp = Kp + (size_t)(jb + jf * 16 + l16) * HEAD_DIM;
            v16bf kb0 = *(const v16bf*)(kp + base16);
            v16bf kb1 = *(const v16bf*)(kp + 32 + base16);
            s[jf] = __builtin_amdgcn_wmma_f32_16x16x32_bf16(
                false, qa[0].v, false, kb0, (short)0, s[jf], false, false);
            s[jf] = __builtin_amdgcn_wmma_f32_16x16x32_bf16(
                false, qa[1].v, false, kb1, (short)0, s[jf], false, false);
        }
        // online softmax; row = r + half*8 lives across one 16-lane half
#pragma unroll
        for (int r = 0; r < 8; ++r) {
            float bm = -1e30f;
#pragma unroll
            for (int jf = 0; jf < 4; ++jf) {
                s[jf][r] *= ATT_SCALE;
                bm = fmaxf(bm, s[jf][r]);
            }
            for (int off = 1; off < 16; off <<= 1)
                bm = fmaxf(bm, __shfl_xor(bm, off, 32));
            float mn   = fmaxf(mrow[r], bm);
            float corr = __expf(mrow[r] - mn);
            mrow[r] = mn;
            float rs = 0.f;
#pragma unroll
            for (int jf = 0; jf < 4; ++jf) {
                float pv = __expf(s[jf][r] - mn);
                s[jf][r] = pv;
                rs += pv;
            }
            for (int off = 1; off < 16; off <<= 1)
                rs += __shfl_xor(rs, off, 32);
            lrow[r] = lrow[r] * corr + rs;
#pragma unroll
            for (int of = 0; of < 4; ++of) o[of][r] *= corr;
        }
        // C-layout -> LDS (bf16), then reload in A-layout (wave-private region)
#pragma unroll
        for (int jf = 0; jf < 4; ++jf)
#pragma unroll
            for (int r = 0; r < 8; ++r)
                plds[wave][r + half * 8][jf * 16 + l16] = (bf16_t)s[jf][r];
        asm volatile("s_wait_dscnt 0" ::: "memory");
#pragma unroll
        for (int ks = 0; ks < 2; ++ks) {
            FragA pa;
            const bf16_t* lp = &plds[wave][l16][ks * 32];
            pa.h[0] = *(const v8bf*)(lp + base16);
            pa.h[1] = *(const v8bf*)(lp + base16 + 8);
#pragma unroll
            for (int of = 0; of < 4; ++of) {
                const bf16_t* vp =
                    Vp + (size_t)(of * 16 + l16) * SEQ + jb + ks * 32 + base16;
                v16bf vb = *(const v16bf*)vp;
                o[of] = __builtin_amdgcn_wmma_f32_16x16x32_bf16(
                    false, pa.v, false, vb, (short)0, o[of], false, false);
            }
        }
    }
    // normalize, store as [b, n, h*64+d] bf16 for the output GEMM
    const int b = bh / HEADS, h = bh % HEADS;
#pragma unroll
    for (int r = 0; r < 8; ++r) {
        float inv = 1.f / lrow[r];
        int row = qbase + r + half * 8;
        size_t obase = ((size_t)(b * SEQ + row)) * DIMX + h * HEAD_DIM;
#pragma unroll
        for (int of = 0; of < 4; ++of)
            AO[obase + of * 16 + l16] = (bf16_t)(o[of][r] * inv);
    }
}

// ---------------------------------------------------------------- output GEMM
// out[m, e] = sum_k AO[m,k] * Wout[e,k] + bias[e];  M=8192, N=1024, K=1024
__global__ __launch_bounds__(128) void out_gemm_kernel(
    const bf16_t* __restrict__ AO, const bf16_t* __restrict__ W,
    const float* __restrict__ bias, float* __restrict__ out) {
    const int lane  = threadIdx.x & 31;
    const int wave  = threadIdx.x >> 5;
    const int tileM = blockIdx.x * 64;
    const int tileN = blockIdx.y * 256 + wave * 64;
    const int half  = lane >> 4;
    const int l16   = lane & 15;
    const int base8  = half * 8;
    const int base16 = half * 16;

    v8f acc[4][4] = {};
    for (int k = 0; k < DIMX; k += 32) {
        FragA a[4];
#pragma unroll
        for (int mi = 0; mi < 4; ++mi) {
            const bf16_t* p = AO + (size_t)(tileM + mi * 16 + l16) * DIMX + k + base16;
            a[mi].h[0] = *(const v8bf*)p;
            a[mi].h[1] = *(const v8bf*)(p + 8);
        }
        v16bf b[4];
#pragma unroll
        for (int ni = 0; ni < 4; ++ni) {
            const bf16_t* p = W + (size_t)(tileN + ni * 16 + l16) * DIMX + k + base16;
            b[ni] = *(const v16bf*)p;
        }
#pragma unroll
        for (int mi = 0; mi < 4; ++mi)
#pragma unroll
            for (int ni = 0; ni < 4; ++ni)
                acc[mi][ni] = __builtin_amdgcn_wmma_f32_16x16x32_bf16(
                    false, a[mi].v, false, b[ni], (short)0, acc[mi][ni], false, false);
    }
#pragma unroll
    for (int mi = 0; mi < 4; ++mi) {
#pragma unroll
        for (int ni = 0; ni < 4; ++ni) {
            int e = tileN + ni * 16 + l16;
            float bv = bias[e];
#pragma unroll
            for (int r = 0; r < 8; ++r) {
                int m = tileM + mi * 16 + r + half * 8;
                out[(size_t)m * DIMX + e] = acc[mi][ni][r] + bv;
            }
        }
    }
}

// ---------------------------------------------------------------- launch
extern "C" void kernel_launch(void* const* d_in, const int* in_sizes, int n_in,
                              void* d_out, int out_size, void* d_ws, size_t ws_size,
                              hipStream_t stream) {
    const float* x     = (const float*)d_in[0];
    const float* w_qkv = (const float*)d_in[1];
    const float* w_out = (const float*)d_in[2];
    const float* b_out = (const float*)d_in[3];
    float* out = (float*)d_out;

    char* ws = (char*)d_ws;
    bf16_t* xbf   = (bf16_t*)ws; ws += (size_t)MTOT * DIMX * sizeof(bf16_t);
    bf16_t* wqbf  = (bf16_t*)ws; ws += (size_t)QKV_N * DIMX * sizeof(bf16_t);
    bf16_t* wobf  = (bf16_t*)ws; ws += (size_t)DIMX * DIMX * sizeof(bf16_t);
    bf16_t* Qb    = (bf16_t*)ws; ws += (size_t)MTOT * DIMX * sizeof(bf16_t);
    bf16_t* Kb    = (bf16_t*)ws; ws += (size_t)MTOT * DIMX * sizeof(bf16_t);
    bf16_t* Vtb   = (bf16_t*)ws; ws += (size_t)MTOT * DIMX * sizeof(bf16_t);
    bf16_t* AOb   = (bf16_t*)ws; ws += (size_t)MTOT * DIMX * sizeof(bf16_t);

    f32_to_bf16_kernel<<<2048, 256, 0, stream>>>(x, xbf, MTOT * DIMX);
    f32_to_bf16_kernel<<<2048, 256, 0, stream>>>(w_qkv, wqbf, QKV_N * DIMX);
    f32_to_bf16_kernel<<<1024, 256, 0, stream>>>(w_out, wobf, DIMX * DIMX);

    qkv_gemm_kernel<<<dim3(MTOT / 64, QKV_N / 256), 128, 0, stream>>>(
        xbf, wqbf, Qb, Kb, Vtb);
    attn_kernel<<<dim3(SEQ / 64, BATCH * HEADS), 128, 0, stream>>>(
        Qb, Kb, Vtb, AOb);
    out_gemm_kernel<<<dim3(MTOT / 64, DIMX / 256), 128, 0, stream>>>(
        AOb, wobf, b_out, out);
}
